// SpatialTemporalConv_74431783240196
// MI455X (gfx1250) — compile-verified, matrix operand order checked by reading, not run on hardware
//
#include <hip/hip_runtime.h>

// ---- problem constants ----
constexpr int kS    = 288;        // SEQ_LEN
constexpr int kF    = 16;         // IN_VAR
constexpr int kB    = 64 * 134;   // 8576 batch items
constexpr int kNE   = 96;         // edges
constexpr int kSH   = 296;        // half-stride for row-major bf16 [16][kSH]
constexpr int kNT   = 18;         // 288/16 tiles
constexpr int kMB   = 2;          // batch items per block (weight-fragment reuse)
constexpr int kTile = kS * kF;    // 4608 floats per batch item

typedef __bf16 bh;
typedef __bf16 bh4   __attribute__((ext_vector_type(4)));
typedef __bf16 bh8   __attribute__((ext_vector_type(8)));
typedef __bf16 v16bf __attribute__((ext_vector_type(16)));
typedef float  v8f   __attribute__((ext_vector_type(8)));
typedef float  f32x4 __attribute__((ext_vector_type(4)));
typedef unsigned int u32x4 __attribute__((ext_vector_type(4)));
typedef int i32x4 __attribute__((ext_vector_type(4)));
typedef int i32x8 __attribute__((ext_vector_type(8)));

#if defined(__HIP_DEVICE_COMPILE__) && __has_builtin(__builtin_amdgcn_tensor_load_to_lds)
#define HAVE_TDM 1
#else
#define HAVE_TDM 0
#endif

static __device__ inline v16bf ldfrag(const bh* p0, const bh* p1) {
  union { v16bf v; bh8 h[2]; } u;
  u.h[0] = *(const bh8*)p0;
  u.h[1] = *(const bh8*)p1;
  return u.v;
}
#define WMMA_BF16(a, b, c) \
  __builtin_amdgcn_wmma_f32_16x16x32_bf16(false, (a), false, (b), (short)0, (c), false, false)

// ---------------- prep kernel 1: tiny constant matrices ----------------
__global__ __launch_bounds__(128) void stc_prep_small(
    const int* __restrict__ ei, const float* __restrict__ c1w,
    const float* __restrict__ c2w, const float* __restrict__ fcw,
    bh* __restrict__ WcA, bh* __restrict__ FcA, bh* __restrict__ M16A,
    float* __restrict__ sumW)
{
  __shared__ int sdeg[16];
  const int tid = threadIdx.x;
  if (tid < 16) sdeg[tid] = 0;
  __syncthreads();
  if (tid < kNE) atomicAdd(&sdeg[ei[kNE + tid]], 1);
  __syncthreads();

  for (int idx = tid; idx < 16 * 32; idx += 128) {     // M16A (K padded to 32)
    int n = idx >> 5, k = idx & 31;
    float v = 0.0f;
    if (k < 16) {
      int cnt = 0;
      for (int e = 0; e < kNE; ++e) cnt += (ei[kNE + e] == n && ei[e] == k) ? 1 : 0;
      int d = sdeg[n] > 1 ? sdeg[n] : 1;
      v = (float)cnt / (float)d;
    }
    M16A[idx] = (bh)v;
  }
  for (int idx = tid; idx < 16 * 96; idx += 128) {     // WcA: k = tap*16+i, k>=80 zero
    int o = idx / 96, k = idx % 96;
    int tap = k >> 4, i = k & 15;
    float v = 0.0f;
    if (tap < 5) v = c2w[o * 80 + i * 5 + tap];
    if (tap >= 1 && tap <= 3) v += c1w[o * 48 + i * 3 + (tap - 1)];
    WcA[idx] = (bh)v;
  }
  for (int idx = tid; idx < 16 * 32; idx += 128) FcA[idx] = (bh)fcw[idx];
  if (tid < 16) {
    float s = 0.0f;
    for (int k = 0; k < 16; ++k) s += fcw[tid * 32 + k];
    sumW[tid] = s;
  }
}

// ---------------- prep kernel 2: Wl/Wr f32 -> bf16 ----------------
__global__ void stc_prep_w(const float* __restrict__ Wl, const float* __restrict__ Wr,
                           bh* __restrict__ WlB, bh* __restrict__ WrB)
{
  int i = blockIdx.x * 256 + threadIdx.x;
  if (i < kS * kS) { WlB[i] = (bh)Wl[i]; WrB[i] = (bh)Wr[i]; }
}

// ---------------- main fused kernel: 2 batch items per block ----------------
__global__ __launch_bounds__(128) void stc_main(
    const float* __restrict__ src, const float* __restrict__ bgcn,
    const bh* __restrict__ WlB, const bh* __restrict__ WrB,
    const bh* __restrict__ WcA, const bh* __restrict__ FcA,
    const bh* __restrict__ M16A, const float* __restrict__ sumW,
    const float* __restrict__ fcb, float* __restrict__ out)
{
#if HAVE_TDM
  __shared__ __align__(16) float stage[kMB][kTile];   // raw src tiles via TDM
#endif
  __shared__ __align__(16) bh sXh[kMB][16][kSH];      // x2 row-major  [f][s]
  __shared__ __align__(16) bh sXsi[kMB][292][16];     // x2 col-major  [s+2][f], halo zero
  __shared__ __align__(16) bh sMh[kMB][16][kSH];      // mean row-major
  __shared__ __align__(16) bh sWc[16][96];
  __shared__ __align__(16) bh sFc[16][32];
  __shared__ __align__(16) bh sM16[16][32];
  __shared__ float sSumW[16];
  __shared__ float sFcB[16];

  const int b0   = blockIdx.x * kMB;
  const int tid  = threadIdx.x;
  const int lane = tid & 31;
  const int wave = __builtin_amdgcn_readfirstlane(tid >> 5);  // scalar -> uniform branches
  const int rlo  = lane & 15;
  const int hi8  = (lane >= 16) ? 8 : 0;
  const float* srcb0 = src + (size_t)b0 * kTile;

  // ---- stage 0a: kick off TDM DMA of both src tiles (36864 B, contiguous) ----
  const float* tilePtr;
#if HAVE_TDM
  if (wave == 0) {
    u32x4 g0; i32x8 g1;
    i32x4 g2 = {0, 0, 0, 0}, g3 = {0, 0, 0, 0};
    i32x8 g4 = {0, 0, 0, 0, 0, 0, 0, 0};
    unsigned ldsOff = (unsigned)(uintptr_t)&stage[0][0];
    unsigned long long ga = (unsigned long long)(uintptr_t)srcb0;
    g0[0] = 1u;                                           // count=1 valid descriptor
    g0[1] = ldsOff;                                       // lds_addr
    g0[2] = (unsigned)ga;                                 // global_addr[31:0]
    g0[3] = (unsigned)((ga >> 32) & 0x1ffffffu) | 0x80000000u; // addr[56:32] | type=2
    g1[0] = 0x20000;                                      // data_size = 4 bytes
    g1[1] = (int)((unsigned)(kMB * kTile) << 16);         // tensor_dim0 lo16
    g1[2] = (int)(1u << 16);                              // dim0 hi16=0 | tensor_dim1=1
    g1[3] = (int)((unsigned)(kMB * kTile) << 16);         // dim1 hi=0 | tile_dim0
    g1[4] = 0;                                            // tile_dim1/2 unused
    g1[5] = kMB * kTile;                                  // tensor_dim0_stride
    g1[6] = 0; g1[7] = 0;
    __builtin_amdgcn_tensor_load_to_lds(g0, g1, g2, g3, g4, 0);  // 6-arg form (clang-23)
  }
  tilePtr = &stage[0][0];
#else
  tilePtr = srcb0;
#endif

  // ---- stage 0b: stage constants, zero halos (overlaps DMA) ----
  for (int i = tid; i < 16 * 96; i += 128) (&sWc[0][0])[i] = WcA[i];
  for (int i = tid; i < 16 * 32; i += 128) { (&sFc[0][0])[i] = FcA[i]; (&sM16[0][0])[i] = M16A[i]; }
  if (tid < 16) { sSumW[tid] = sumW[tid]; sFcB[tid] = fcb[tid]; }
  {
    int mb = tid >> 6, j = tid & 63;     // 128 threads -> 2 mb x 64 halo halfs
    int r = (j < 32) ? (j >> 4) : (290 + ((j - 32) >> 4));
    sXsi[mb][r][j & 15] = (bh)0.0f;
  }
#if HAVE_TDM
  if (wave == 0) __builtin_amdgcn_s_wait_tensorcnt(0);
#endif
  __syncthreads();

  // ---- stage 0c: convert f32 tiles -> bf16 row-major + col-major ----
  for (int idx = tid; idx < kMB * kTile / 4; idx += 128) {
    int mb = (idx >= kTile / 4) ? 1 : 0;
    int rem = idx - mb * (kTile / 4);
    int s = rem >> 2, f0 = (rem & 3) << 2;
    f32x4 v = *(const f32x4*)(tilePtr + idx * 4);
    bh v0 = (bh)v.x, v1 = (bh)v.y, v2 = (bh)v.z, v3 = (bh)v.w;
    bh4 p = {v0, v1, v2, v3};
    *(bh4*)&sXsi[mb][s + 2][f0] = p;
    sXh[mb][f0 + 0][s] = v0; sXh[mb][f0 + 1][s] = v1;
    sXh[mb][f0 + 2][s] = v2; sXh[mb][f0 + 3][s] = v3;
  }
  __syncthreads();

  // ---- stage 1: mean = M16 @ X (A shared across mb) ----
  {
    v16bf aM = ldfrag(&sM16[rlo][hi8], &sM16[rlo][16 + hi8]);
    for (int t = wave; t < kNT; t += 4) {
      int s = t * 16 + rlo;
#pragma unroll
      for (int mb = 0; mb < kMB; ++mb) {
        const bh* base = &sXsi[mb][s + 2][0];
        v16bf bB = ldfrag(base + hi8, base + 16 + hi8);
        v8f c = {};
        c = WMMA_BF16(aM, bB, c);
#pragma unroll
        for (int j = 0; j < 8; ++j) sMh[mb][j + hi8][s] = (bh)c[j];
      }
    }
  }
  __syncthreads();

  // ---- stage 2: feat + conv + fc fully in registers per tile ----
  const v16bf aF = ldfrag(&sFc[rlo][hi8], &sFc[rlo][16 + hi8]);
  for (int t = wave; t < kNT; t += 4) {
    const int t0  = t * 16;
    const int col = t0 + rlo;            // output column (t for feat == s for conv/fc)

    // feat: B-fragments (Wl/Wr rows) loaded once, reused for both mb
    v8f cF0 = {}, cF1 = {};
    const bh* wlRow = WlB + (size_t)col * kS;
    const bh* wrRow = WrB + (size_t)col * kS;
    __builtin_prefetch(wlRow, 0, 1);
    __builtin_prefetch(wrRow, 0, 1);
#pragma unroll 3
    for (int k0 = 0; k0 < kS; k0 += 32) {
      int ka = k0 + hi8;
      v16bf bL = ldfrag(wlRow + ka, wlRow + ka + 16);
      v16bf bR = ldfrag(wrRow + ka, wrRow + ka + 16);
      v16bf a0 = ldfrag(&sMh[0][rlo][ka], &sMh[0][rlo][ka + 16]);
      cF0 = WMMA_BF16(a0, bL, cF0);
      v16bf a1 = ldfrag(&sMh[1][rlo][ka], &sMh[1][rlo][ka + 16]);
      cF1 = WMMA_BF16(a1, bL, cF1);
      v16bf x0 = ldfrag(&sXh[0][rlo][ka], &sXh[0][rlo][ka + 16]);
      cF0 = WMMA_BF16(x0, bR, cF0);
      v16bf x1 = ldfrag(&sXh[1][rlo][ka], &sXh[1][rlo][ka + 16]);
      cF1 = WMMA_BF16(x1, bR, cF1);
    }

    // conv: A (Wc) shared across mb; k = tap*16+i, K padded 80->96 (A zero)
    v8f cC0 = {}, cC1 = {};
#pragma unroll
    for (int k0 = 0; k0 < 96; k0 += 32) {
      int ka = k0 + hi8;
      v16bf aW = ldfrag(&sWc[rlo][ka], &sWc[rlo][ka + 16]);
      int tap0 = ka >> 4, i0 = ka & 15;
      int tap1 = (ka + 16) >> 4;
      int r0 = col + tap0; if (r0 > 291) r0 = 291;   // clamp: finite where A==0
      int r1 = col + tap1; if (r1 > 291) r1 = 291;
      v16bf bc0 = ldfrag(&sXsi[0][r0][i0], &sXsi[0][r1][i0]);
      cC0 = WMMA_BF16(aW, bc0, cC0);
      v16bf bc1 = ldfrag(&sXsi[1][r0][i0], &sXsi[1][r1][i0]);
      cC1 = WMMA_BF16(aW, bc1, cC1);
    }

    // fc: D-register layout of feat/conv tiles IS the fc B-fragment layout.
    const int s = col;
    const float g = bgcn[s];
#pragma unroll
    for (int mb = 0; mb < kMB; ++mb) {
      const v8f& f  = mb ? cF1 : cF0;
      const v8f& cc = mb ? cC1 : cC0;
      union { v16bf v; bh8 h[2]; } u;
#pragma unroll
      for (int j = 0; j < 8; ++j) { u.h[0][j] = (bh)f[j]; u.h[1][j] = (bh)cc[j]; }
      v8f d = {};
      d = WMMA_BF16(aF, u.v, d);
      const float* sp = tilePtr + mb * kTile + s * 16 + hi8;   // residual from staged tile
      f32x4 s0 = *(const f32x4*)sp;
      f32x4 s1 = *(const f32x4*)(sp + 4);
      f32x4 r0, r1;
      r0.x = s0.x + d[0] + sFcB[hi8 + 0] + g * sSumW[hi8 + 0];
      r0.y = s0.y + d[1] + sFcB[hi8 + 1] + g * sSumW[hi8 + 1];
      r0.z = s0.z + d[2] + sFcB[hi8 + 2] + g * sSumW[hi8 + 2];
      r0.w = s0.w + d[3] + sFcB[hi8 + 3] + g * sSumW[hi8 + 3];
      r1.x = s1.x + d[4] + sFcB[hi8 + 4] + g * sSumW[hi8 + 4];
      r1.y = s1.y + d[5] + sFcB[hi8 + 5] + g * sSumW[hi8 + 5];
      r1.z = s1.z + d[6] + sFcB[hi8 + 6] + g * sSumW[hi8 + 6];
      r1.w = s1.w + d[7] + sFcB[hi8 + 7] + g * sSumW[hi8 + 7];
      float* op = out + (size_t)(b0 + mb) * kTile + s * 16 + hi8;
      *(f32x4*)op       = r0;
      *(f32x4*)(op + 4) = r1;
    }
  }
}

extern "C" void kernel_launch(void* const* d_in, const int* in_sizes, int n_in,
                              void* d_out, int out_size, void* d_ws, size_t ws_size,
                              hipStream_t stream) {
  const float* src  = (const float*)d_in[0];
  const int*   ei   = (const int*)d_in[1];
  const float* Wl   = (const float*)d_in[2];
  const float* Wr   = (const float*)d_in[3];
  const float* bgcn = (const float*)d_in[4];
  const float* c1w  = (const float*)d_in[5];
  const float* c2w  = (const float*)d_in[6];
  const float* fcw  = (const float*)d_in[7];
  const float* fcb  = (const float*)d_in[8];
  float* out = (float*)d_out;

  char* ws = (char*)d_ws;
  bh*    WlB  = (bh*)(ws);                       // 288*288*2 = 165888
  bh*    WrB  = (bh*)(ws + 165888);              // 165888
  bh*    WcA  = (bh*)(ws + 331776);              // 3072
  bh*    FcA  = (bh*)(ws + 334848);              // 1024
  bh*    M16A = (bh*)(ws + 335872);              // 1024
  float* sumW = (float*)(ws + 336896);           // 64

  stc_prep_small<<<1, 128, 0, stream>>>(ei, c1w, c2w, fcw, WcA, FcA, M16A, sumW);
  stc_prep_w<<<(kS * kS + 255) / 256, 256, 0, stream>>>(Wl, Wr, WlB, WrB);
  stc_main<<<kB / kMB, 128, 0, stream>>>(src, bgcn, WlB, WrB, WcA, FcA, M16A, sumW, fcb, out);
}
